// QuantLinear_25091198943628
// MI455X (gfx1250) — 8-bit matrix operand order repaired and checked by reading, compiled, not run on hardware
//
#include <hip/hip_runtime.h>
#include <hip/hip_bf16.h>
#include <stdint.h>

typedef int   v8i __attribute__((ext_vector_type(8)));
typedef float v8f __attribute__((ext_vector_type(8)));

#define M_DIM 512
#define K_DIM 8192
#define N_DIM 14336
#define KSTEPS 128
#define ZP 8

#define A_STRIDE 80
#define B_STRIDE 80

#if __has_builtin(__builtin_amdgcn_global_load_async_to_lds_b128)
#define HAVE_ASYNC_LDS 1
#else
#define HAVE_ASYNC_LDS 0
#endif

__device__ __forceinline__ void stage16(const unsigned char* g, unsigned char* l) {
#if HAVE_ASYNC_LDS
  typedef int i32x4 __attribute__((vector_size(16)));
  typedef __attribute__((address_space(1))) i32x4* gp_t;
  typedef __attribute__((address_space(3))) i32x4* lp_t;
  __builtin_amdgcn_global_load_async_to_lds_b128(
      (gp_t)(const __attribute__((address_space(1))) void*)g,
      (lp_t)(__attribute__((address_space(3))) void*)l, 0, 0);
#else
  *(uint4*)l = *(const uint4*)g;
#endif
}

__device__ __forceinline__ void wait_async_done() {
#if HAVE_ASYNC_LDS
#if __has_builtin(__builtin_amdgcn_s_wait_asynccnt)
  __builtin_amdgcn_s_wait_asynccnt(0);
#else
  asm volatile("s_wait_asynccnt 0x0" ::: "memory");
#endif
#endif
}

__global__ __launch_bounds__(256) void quant_rows(const float* __restrict__ A,
                                                  unsigned char* __restrict__ qA,
                                                  float* __restrict__ s1) {
  __shared__ float red[256];
  const int row = blockIdx.x;
  const int t = threadIdx.x;
  const float* arow = A + (size_t)row * K_DIM;

  float4 v[8];
  float mx = 0.f;
#pragma unroll
  for (int c = 0; c < 8; ++c) {
    v[c] = *(const float4*)(arow + c * 1024 + t * 4);
    mx = fmaxf(mx, fmaxf(fmaxf(fabsf(v[c].x), fabsf(v[c].y)),
                         fmaxf(fabsf(v[c].z), fabsf(v[c].w))));
  }
  red[t] = mx;
  __syncthreads();
  for (int s = 128; s > 0; s >>= 1) {
    if (t < s) red[t] = fmaxf(red[t], red[t + s]);
    __syncthreads();
  }
  const float m = red[0];
  const float scale = m * (1.f / 127.f);
  const float inv = (m > 0.f) ? (127.f / m) : 0.f;
  if (t == 0) s1[row] = scale;

#pragma unroll
  for (int c = 0; c < 8; ++c) {
    int q0 = (int)fminf(fmaxf(rintf(v[c].x * inv), -128.f), 127.f);
    int q1 = (int)fminf(fmaxf(rintf(v[c].y * inv), -128.f), 127.f);
    int q2 = (int)fminf(fmaxf(rintf(v[c].z * inv), -128.f), 127.f);
    int q3 = (int)fminf(fmaxf(rintf(v[c].w * inv), -128.f), 127.f);
    uint32_t p = (uint32_t)(q0 & 0xFF) | ((uint32_t)(q1 & 0xFF) << 8) |
                 ((uint32_t)(q2 & 0xFF) << 16) | ((uint32_t)(q3 & 0xFF) << 24);
    *(uint32_t*)(qA + (size_t)row * K_DIM + c * 1024 + t * 4) = p;
  }
}

__global__ __launch_bounds__(256) void w4a8_gemm(
    const unsigned char* __restrict__ qA, const float* __restrict__ s1,
    const int* __restrict__ w4, const float* __restrict__ s_group,
    const float* __restrict__ s_channel, const float* __restrict__ bias,
    float* __restrict__ out) {
  __shared__ __align__(16) unsigned char sA[2][128 * A_STRIDE];
  __shared__ __align__(16) unsigned char sB[2][64 * B_STRIDE];
  __shared__ float sS1[128];

  const int t = threadIdx.x;
  const int lane = t & 31;
  const int wave = t >> 5;
  const int wm = wave & 3;
  const int wn = wave >> 2;
  const int m0 = blockIdx.x * 128;
  const int n0 = blockIdx.y * 64;

  if (t < 128) sS1[t] = s1[m0 + t];

  const int a_row = t >> 1;
  const int a_col = (t & 1) * 32;
  const int b_n = t & 63;
  const int b_q0 = (t >> 6) * 4;

  const int lrow = lane & 15;
  const int lhi = lane >> 4;

  const unsigned char* Ab = qA + (size_t)m0 * K_DIM + (size_t)a_row * K_DIM + a_col;
  unsigned char* const lA0 = &sA[0][a_row * A_STRIDE + a_col];
  unsigned char* const lA1 = &sA[1][a_row * A_STRIDE + a_col];
  const int* const wb = w4 + (size_t)n0 + b_n;
  unsigned char* const lB0 = &sB[0][b_n * B_STRIDE];
  unsigned char* const lB1 = &sB[1][b_n * B_STRIDE];

  v8f facc[2][2] = {};
  v8i iacc[2][2] = {};
  const v8i vzero = {};

  {
    stage16(Ab, lA0);
    stage16(Ab + 16, lA0 + 16);
#pragma unroll
    for (int q = 0; q < 4; ++q) {
      const int k = (b_q0 + q) * 4;
      const int* wp = wb + (size_t)k * N_DIM;
      const int w0 = wp[0];
      const int w1 = wp[N_DIM];
      const int w2 = wp[2 * N_DIM];
      const int w3 = wp[3 * N_DIM];
      const uint32_t p = (uint32_t)((w0 - ZP) & 0xFF) |
                         ((uint32_t)((w1 - ZP) & 0xFF) << 8) |
                         ((uint32_t)((w2 - ZP) & 0xFF) << 16) |
                         ((uint32_t)((w3 - ZP) & 0xFF) << 24);
      *(uint32_t*)(lB0 + k) = p;
    }
  }

#pragma unroll 1
  for (int ks = 0; ks < KSTEPS; ++ks) {
    const int cur = ks & 1;
    wait_async_done();
    __syncthreads();

    if (ks + 1 < KSTEPS) {
      const int k0n = (ks + 1) * 64;
      unsigned char* lA = cur ? lA0 : lA1;
      unsigned char* lB = cur ? lB0 : lB1;
      stage16(Ab + k0n, lA);
      stage16(Ab + k0n + 16, lA + 16);
#pragma unroll
      for (int q = 0; q < 4; ++q) {
        const int k = (b_q0 + q) * 4;
        const int* wp = wb + (size_t)(k0n + k) * N_DIM;
        if (ks + 2 < KSTEPS) __builtin_prefetch(wp + (size_t)64 * N_DIM, 0, 1);
        const int w0 = wp[0];
        const int w1 = wp[N_DIM];
        const int w2 = wp[2 * N_DIM];
        const int w3 = wp[3 * N_DIM];
        const uint32_t p = (uint32_t)((w0 - ZP) & 0xFF) |
                           ((uint32_t)((w1 - ZP) & 0xFF) << 8) |
                           ((uint32_t)((w2 - ZP) & 0xFF) << 16) |
                           ((uint32_t)((w3 - ZP) & 0xFF) << 24);
        *(uint32_t*)(lB + k) = p;
      }
    }

    v8i Af[2], Bf[2];
#pragma unroll
    for (int i = 0; i < 2; ++i) {
      const unsigned char* base =
          &sA[cur][(wm * 32 + i * 16 + lrow) * A_STRIDE + lhi * 16];
      const uint2 a0 = *(const uint2*)(base);
      const uint2 a1 = *(const uint2*)(base + 8);
      const uint2 a2 = *(const uint2*)(base + 32);
      const uint2 a3 = *(const uint2*)(base + 40);
      Af[i][0] = a0.x; Af[i][1] = a0.y;
      Af[i][2] = a1.x; Af[i][3] = a1.y;
      Af[i][4] = a2.x; Af[i][5] = a2.y;
      Af[i][6] = a3.x; Af[i][7] = a3.y;
    }
#pragma unroll
    for (int j = 0; j < 2; ++j) {
      const unsigned char* base =
          &sB[cur][(wn * 32 + j * 16 + lrow) * B_STRIDE + lhi * 16];
      const uint4 b0 = *(const uint4*)(base);
      const uint4 b1 = *(const uint4*)(base + 32);
      Bf[j][0] = b0.x; Bf[j][1] = b0.y; Bf[j][2] = b0.z; Bf[j][3] = b0.w;
      Bf[j][4] = b1.x; Bf[j][5] = b1.y; Bf[j][6] = b1.z; Bf[j][7] = b1.w;
    }

#pragma unroll
    for (int i = 0; i < 2; ++i)
#pragma unroll
      for (int j = 0; j < 2; ++j)
        iacc[i][j] = __builtin_amdgcn_wmma_i32_16x16x64_iu8(
             true, Af[i],  true, Bf[j], iacc[i][j],
             false,  false);

    if (ks & 1) {
      const int g = ks >> 1;
#pragma unroll
      for (int j = 0; j < 2; ++j) {
        const float sg = s_group[(size_t)g * N_DIM + n0 + wn * 32 + j * 16 + lrow];
#pragma unroll
        for (int i = 0; i < 2; ++i) {
#pragma unroll
          for (int v = 0; v < 8; ++v)
            facc[i][j][v] += (float)iacc[i][j][v] * sg;
          iacc[i][j] = vzero;
        }
      }
    }
  }

#pragma unroll
  for (int j = 0; j < 2; ++j) {
    const int n = n0 + wn * 32 + j * 16 + lrow;
    const float sc = s_channel[n];
    const float bb = bias[n];
#pragma unroll
    for (int i = 0; i < 2; ++i) {
      const int mb = wm * 32 + i * 16 + lhi * 8;
#pragma unroll
      for (int v = 0; v < 8; ++v) {
        const int m = mb + v;
        out[(size_t)(m0 + m) * N_DIM + n] = facc[i][j][v] * sS1[m] * sc + bb;
      }
    }
  }
}

extern "C" void kernel_launch(void* const* d_in, const int* in_sizes, int n_in,
                              void* d_out, int out_size, void* d_ws, size_t ws_size,
                              hipStream_t stream) {
  const float* A = (const float*)d_in[0];
  const int* w4 = (const int*)d_in[1];
  const float* s_group = (const float*)d_in[2];
  const float* s_channel = (const float*)d_in[3];
  const float* bias = (const float*)d_in[4];
  float* out = (float*)d_out;

  unsigned char* qA = (unsigned char*)d_ws;
  float* s1 = (float*)((char*)d_ws + (size_t)M_DIM * K_DIM);

  quant_rows<<<M_DIM, 256, 0, stream>>>(A, qA, s1);

  dim3 grid(M_DIM / 128, N_DIM / 64);
  w4a8_gemm<<<grid, 256, 0, stream>>>(qA, s1, w4, s_group, s_channel, bias, out);
}
